// ObjectRelationModule_11579231830247
// MI455X (gfx1250) — matrix operand order repaired and checked by reading, not compiled, not run on hardware
//
#include <hip/hip_runtime.h>

// ---------------------------------------------------------------------------
// CDNA5 (gfx1250) wave32 WMMA helpers: v_wmma_f32_16x16x32_bf16
// Fragment layouts per CDNA5 ISA 7.12.2 (16-bit A 16x32, B 32x16, f32 C/D).
//   A elem i  -> K = (i<8) ? half*8+i : 16+half*8+(i-8)   (two contiguous runs)
//   B elem i  -> K = same as A                              (two contiguous runs)
//   C/D vgpr r-> M = r + 8*half, N = lane%16
// ---------------------------------------------------------------------------
typedef __attribute__((ext_vector_type(16))) __bf16 v16bf;
typedef __attribute__((ext_vector_type(8)))  __bf16 v8bf;
typedef __attribute__((ext_vector_type(8)))  float  v8f;

__device__ __forceinline__ v8f wmma_bf16(v16bf a, v16bf b, v8f c) {
  return __builtin_amdgcn_wmma_f32_16x16x32_bf16(false, a, false, b,
                                                 (short)0, c, false, false);
}
__device__ __forceinline__ v16bf cat8(v8bf a, v8bf b) {
  return __builtin_shufflevector(a, b, 0,1,2,3,4,5,6,7,8,9,10,11,12,13,14,15);
}
// pack 4 float4 (16 contiguous-ordered floats) into a bf16 fragment vector
__device__ __forceinline__ v16bf pack16(float4 a, float4 b, float4 c, float4 d) {
  v16bf r;
  r[0]  = (__bf16)a.x; r[1]  = (__bf16)a.y; r[2]  = (__bf16)a.z; r[3]  = (__bf16)a.w;
  r[4]  = (__bf16)b.x; r[5]  = (__bf16)b.y; r[6]  = (__bf16)b.z; r[7]  = (__bf16)b.w;
  r[8]  = (__bf16)c.x; r[9]  = (__bf16)c.y; r[10] = (__bf16)c.z; r[11] = (__bf16)c.w;
  r[12] = (__bf16)d.x; r[13] = (__bf16)d.y; r[14] = (__bf16)d.z; r[15] = (__bf16)d.w;
  return r;
}

#define NTOK 1600
#define CCH  256
#define DK   64
#define WGRID 40
#define SLAB_LD 1604   // floats; 16B-aligned rows; (SLAB_LD/4)=401 exact for float4
#define BPAD 40        // bf16 K-stride for LDS tiles (80B, 16B aligned, no conflicts)

// ---------------------------------------------------------------------------
// Kernel 1: geometric relation table.
// w_g_mean[n,m] depends only on (|dy|,|dx|) -> 40x40 table (2.56M -> 1600 evals).
// ---------------------------------------------------------------------------
__global__ void geo_table_kernel(const float* __restrict__ Wg1, const float* __restrict__ bg1,
                                 const float* __restrict__ Wg2, const float* __restrict__ bg2,
                                 const float* __restrict__ Wgr, const float* __restrict__ bgr,
                                 float* __restrict__ wtab) {
  __shared__ float sWg1[4 * 64], sbg1[64], sWg2[64 * 64], sbg2[64], sWgr[64 * 16], sbgr[16];
  int tid = threadIdx.x;
  for (int i = tid; i < 4 * 64; i += blockDim.x) sWg1[i] = Wg1[i];
  for (int i = tid; i < 64; i += blockDim.x) { sbg1[i] = bg1[i]; sbg2[i] = bg2[i]; }
  for (int i = tid; i < 64 * 64; i += blockDim.x) sWg2[i] = Wg2[i];
  for (int i = tid; i < 64 * 16; i += blockDim.x) sWgr[i] = Wgr[i];
  for (int i = tid; i < 16; i += blockDim.x) sbgr[i] = bgr[i];
  __syncthreads();
  int idx = blockIdx.x * blockDim.x + tid;
  if (idx >= WGRID * WGRID) return;
  float adx = (float)(idx % WGRID);
  float ady = (float)(idx / WGRID);
  float g0 = __logf(adx + 1e-8f);
  float g1 = __logf(ady + 1e-8f);
  float g23 = __logf(1e-8f);            // w,h diffs are identically zero
  float e1[64];
#pragma unroll 4
  for (int d = 0; d < 64; ++d) {
    float s = sbg1[d] + g0 * sWg1[0 * 64 + d] + g1 * sWg1[1 * 64 + d]
            + g23 * (sWg1[2 * 64 + d] + sWg1[3 * 64 + d]);
    e1[d] = fmaxf(s, 0.f);
  }
  float e2[64];
  for (int d = 0; d < 64; ++d) {
    float s = sbg2[d];
#pragma unroll 8
    for (int c = 0; c < 64; ++c) s = fmaf(e1[c], sWg2[c * 64 + d], s);
    e2[d] = fmaxf(s, 0.f);
  }
  float acc = 0.f;
  for (int r = 0; r < 16; ++r) {
    float s = sbgr[r];
#pragma unroll 8
    for (int c = 0; c < 64; ++c) s = fmaf(e2[c], sWgr[c * 16 + r], s);
    acc += fmaxf(s, 0.f);
  }
  wtab[idx] = acc * (1.0f / 16.0f);
}

// ---------------------------------------------------------------------------
// Kernel 2: fused QKV projection -> bf16 workspace.
//   q, k: token-major [6400][64]; v: TRANSPOSED channel-major [256][6400]
//   so the attention kernel's B-fragments are contiguous global b128 loads.
// LDS tiles staged as bf16 (convert once at stage time).
// ---------------------------------------------------------------------------
__global__ void qkv_kernel(const float* __restrict__ x,
                           const float* __restrict__ Wq, const float* __restrict__ bq,
                           const float* __restrict__ Wk, const float* __restrict__ bk,
                           const float* __restrict__ Wv, const float* __restrict__ bv,
                           __bf16* __restrict__ qo, __bf16* __restrict__ ko,
                           __bf16* __restrict__ vt) {
  __shared__ __bf16 aT[16 * BPAD];        // [token][Kchunk]
  __shared__ __bf16 wT[384 * BPAD];       // [combined col][Kchunk]
  int tid  = threadIdx.x;
  int lane = tid & 31, wave = tid >> 5;
  int half = lane >> 4, nn = lane & 15;
  int tile = blockIdx.x;                  // global 16-token tile, 0..399
  int b    = tile / 100;
  int n0   = (tile % 100) * 16;           // token base within batch

  v8f acc[6];
#pragma unroll
  for (int s = 0; s < 6; ++s) acc[s] = v8f{};

  for (int kb = 0; kb < CCH; kb += 32) {
    // stage A transposed (bf16): channel row -> contiguous 16 tokens
    {
      int row = tid >> 2, quad = tid & 3;       // row: channel 0..31
      const float4 src = *reinterpret_cast<const float4*>(
          x + ((size_t)(b * CCH + kb + row) * NTOK) + n0 + quad * 4);
      aT[(quad * 4 + 0) * BPAD + row] = (__bf16)src.x;
      aT[(quad * 4 + 1) * BPAD + row] = (__bf16)src.y;
      aT[(quad * 4 + 2) * BPAD + row] = (__bf16)src.z;
      aT[(quad * 4 + 3) * BPAD + row] = (__bf16)src.w;
    }
    // stage weights transposed (bf16): combined cols 0..383 (q|k|v), branchless
    for (int t = tid; t < 32 * 96; t += 128) {
      int row = t / 96, c0 = (t % 96) * 4;
      const float* base = (c0 < 64) ? Wq : ((c0 < 128) ? Wk : Wv);
      int woff = (c0 < 64) ? c0 : ((c0 < 128) ? c0 - 64 : c0 - 128);
      int ld   = (c0 < 128) ? 64 : CCH;
      float4 w4 = *reinterpret_cast<const float4*>(base + (size_t)(kb + row) * ld + woff);
      wT[(c0 + 0) * BPAD + row] = (__bf16)w4.x;
      wT[(c0 + 1) * BPAD + row] = (__bf16)w4.y;
      wT[(c0 + 2) * BPAD + row] = (__bf16)w4.z;
      wT[(c0 + 3) * BPAD + row] = (__bf16)w4.w;
    }
    __syncthreads();
    // A fragment: two 8-bf16 runs -> 2x ds_load_b128, no converts
    const v8bf* ap = reinterpret_cast<const v8bf*>(aT + nn * BPAD + half * 8);
    v16bf af = cat8(ap[0], ap[2]);              // runs at +0 and +16 elems
#pragma unroll
    for (int s = 0; s < 6; ++s) {
      int col0c = (wave * 6 + s) * 16;          // combined col base
      const v8bf* bp = reinterpret_cast<const v8bf*>(wT + (col0c + nn) * BPAD + half * 8);
      v16bf bf = cat8(bp[0], bp[2]);
      acc[s] = wmma_bf16(af, bf, acc[s]);
    }
    __syncthreads();
  }
  // epilogue: +bias, store bf16
#pragma unroll
  for (int s = 0; s < 6; ++s) {
    int ot = wave * 6 + s;
#pragma unroll
    for (int r = 0; r < 8; ++r) {
      int tok = tile * 16 + r + 8 * half;       // C/D: M = r + 8*half, N = nn
      if (ot < 4) {
        int c = ot * 16 + nn;
        qo[(size_t)tok * DK + c] = (__bf16)(acc[s][r] + bq[c]);
      } else if (ot < 8) {
        int c = (ot - 4) * 16 + nn;
        ko[(size_t)tok * DK + c] = (__bf16)(acc[s][r] + bk[c]);
      } else {
        int c = (ot - 8) * 16 + nn;
        vt[(size_t)c * (4 * NTOK) + tok] = (__bf16)(acc[s][r] + bv[c]);
      }
    }
  }
}

// ---------------------------------------------------------------------------
// Kernel 3: fused attention. WG = (batch b, 16-row tile); 16x1600 f32 score
// slab in LDS (320KB/WGP). wtab is copied via ASYNC global->LDS so the copy
// overlaps all of Phase A's WMMA work (completed by s_wait_asynccnt).
// Phase A: QK^T (bf16 K-frags = direct global b128). Phase B: softmax *
// geo-weight, float4-vectorized. Phase C: P@V with V read channel-major
// straight from global (L2-resident) -- no staging, no barriers.
// ---------------------------------------------------------------------------
__global__ void attn_kernel(const float* __restrict__ x,
                            const __bf16* __restrict__ qw, const __bf16* __restrict__ kw,
                            const __bf16* __restrict__ vt, const float* __restrict__ wtab,
                            float* __restrict__ out) {
  extern __shared__ char smem[];
  float*  slab   = (float*)smem;                    // 16 * SLAB_LD
  float*  wtabs  = slab + 16 * SLAB_LD;             // 1600
  float*  red    = wtabs + WGRID * WGRID;           // 16 * 8 partials
  float*  rowmax = red + 16 * 8;                    // 16
  float*  rowZ   = rowmax + 16;                     // 16
  __bf16* qTb    = (__bf16*)(rowZ + 16);            // 16 * 72  [tok][K] bf16

  int tid  = threadIdx.x;
  int lane = tid & 31, wave = tid >> 5;
  int half = lane >> 4, nn = lane & 15;
  int b     = blockIdx.x / 100;
  int rtile = blockIdx.x % 100;
  int rbase = rtile * 16;                           // row-token base within batch

  // Async copy wtab (400 float4) global->LDS; overlaps Phase A, tracked by
  // ASYNCcnt (GLOBAL_LOAD_ASYNC_TO_LDS_B128, CDNA5 ISA 10.7).
  for (int f = tid; f < 400; f += 128) {
    unsigned dstoff = (unsigned)(size_t)(wtabs + f * 4);    // LDS byte address
    const float* src = wtab + f * 4;
    asm volatile("global_load_async_to_lds_b128 %0, %1, off"
                 :: "v"(dstoff), "v"(src) : "memory");
  }
  for (int f = tid; f < 16 * 64; f += 128) {        // q tile, token-major padded
    int m = f >> 6, k = f & 63;
    qTb[m * 72 + k] = qw[(size_t)(b * NTOK + rbase + m) * DK + k];
  }
  __syncthreads();

  // Q fragments (16x64 -> two 16x32 A-frags), direct bf16 b128 LDS reads
  const v8bf* qp = reinterpret_cast<const v8bf*>(qTb + nn * 72 + half * 8);
  v16bf aq0 = cat8(qp[0], qp[2]);                   // K 0..31 runs (+0, +16)
  v16bf aq1 = cat8(qp[4], qp[6]);                   // K 32..63 runs (+32, +48)

  // ---- Phase A: scores = (Q K^T) / sqrt(64) into slab ----
  for (int j = wave; j < 100; j += 4) {
    // B elem i -> K = (i<8) ? half*8+i : 16+half*8+(i-8): two runs in k rows -> 4x global b128
    const v8bf* kt = reinterpret_cast<const v8bf*>(
        kw + (size_t)(b * NTOK + j * 16 + nn) * DK + half * 8);
    v16bf b0 = cat8(kt[0], kt[2]);                  // K 0..31 half
    v16bf b1 = cat8(kt[4], kt[6]);                  // K 32..63 half
    v8f s = v8f{};
    s = wmma_bf16(aq0, b0, s);
    s = wmma_bf16(aq1, b1, s);
#pragma unroll
    for (int r = 0; r < 8; ++r)
      slab[(r + 8 * half) * SLAB_LD + j * 16 + nn] = s[r] * 0.125f;
  }
  asm volatile("s_wait_asynccnt 0x0" ::: "memory"); // wtab copy complete
  __syncthreads();

  // ---- Phase B: softmax * w_g, in place (float4 sweeps) ----
  int row = tid >> 3, seg = tid & 7;                // 16 rows x 8 segs of 200
  float4* slab4 = reinterpret_cast<float4*>(slab + row * SLAB_LD);
  {
    float lmax = -3.0e38f;
    for (int c4 = seg * 50; c4 < seg * 50 + 50; ++c4) {
      float4 v = slab4[c4];
      lmax = fmaxf(lmax, fmaxf(fmaxf(v.x, v.y), fmaxf(v.z, v.w)));
    }
    red[row * 8 + seg] = lmax;
  }
  __syncthreads();
  if (tid < 16) {
    float m = red[tid * 8];
    for (int k = 1; k < 8; ++k) m = fmaxf(m, red[tid * 8 + k]);
    rowmax[tid] = m;
  }
  __syncthreads();
  {
    float m = rowmax[row], lsum = 0.f;
    for (int c4 = seg * 50; c4 < seg * 50 + 50; ++c4) {
      float4 v = slab4[c4];
      v.x = __expf(v.x - m); v.y = __expf(v.y - m);
      v.z = __expf(v.z - m); v.w = __expf(v.w - m);
      slab4[c4] = v;
      lsum += (v.x + v.y) + (v.z + v.w);
    }
    red[row * 8 + seg] = lsum;
  }
  __syncthreads();
  if (tid < 16) {
    float z = 0.f;
    for (int k = 0; k < 8; ++k) z += red[tid * 8 + k];
    rowZ[tid] = z;
  }
  __syncthreads();
  {
    float inv = 1.0f / rowZ[row];
    int ntok = rbase + row, yn = ntok / WGRID, xn = ntok % WGRID;
    for (int ymr = 0; ymr < 5; ++ymr) {             // seg covers 5 grid rows
      int ym = seg * 5 + ymr;
      const float* wrow = wtabs + abs(ym - yn) * WGRID;
      float* srow = slab + row * SLAB_LD + ym * WGRID;
#pragma unroll 4
      for (int xm = 0; xm < WGRID; ++xm)
        srow[xm] *= wrow[abs(xm - xn)] * inv;
    }
  }
  __syncthreads();

  // ---- Phase C: rel = P @ V. V channel-major in global (L2-hot); no LDS,
  // no barriers. Wave owns 4 of 16 channel tiles. ----
  v8f acc[4];
#pragma unroll
  for (int s = 0; s < 4; ++s) acc[s] = v8f{};

  for (int cb = 0; cb < 50; ++cb) {
    // P fragment: two 8-float runs in slab -> 4x ds_load_b128 + cvt
    const float4* p = reinterpret_cast<const float4*>(slab + nn * SLAB_LD + cb * 32 + half * 8);
    v16bf ap = pack16(p[0], p[1], p[4], p[5]);
#pragma unroll
    for (int s = 0; s < 4; ++s) {
      int c = (wave * 4 + s) * 16 + nn;
      const v8bf* vp = reinterpret_cast<const v8bf*>(
          vt + (size_t)c * (4 * NTOK) + b * NTOK + cb * 32 + half * 8);
      v16bf bv = cat8(vp[0], vp[2]);
      acc[s] = wmma_bf16(ap, bv, acc[s]);
    }
  }

  // ---- Epilogue: out[b,c,n] = x[b,c,n] + rel[n,c] (fused transpose) ----
#pragma unroll
  for (int s = 0; s < 4; ++s) {
    int c = (wave * 4 + s) * 16 + nn;
#pragma unroll
    for (int r = 0; r < 8; ++r) {
      int n = rbase + r + 8 * half;
      size_t a = (size_t)(b * CCH + c) * NTOK + n;
      out[a] = x[a] + acc[s][r];
    }
  }
}

// ---------------------------------------------------------------------------
extern "C" void kernel_launch(void* const* d_in, const int* in_sizes, int n_in,
                              void* d_out, int out_size, void* d_ws, size_t ws_size,
                              hipStream_t stream) {
  const float* x   = (const float*)d_in[0];
  const float* Wq  = (const float*)d_in[1];
  const float* bq  = (const float*)d_in[2];
  const float* Wk  = (const float*)d_in[3];
  const float* bk  = (const float*)d_in[4];
  const float* Wv  = (const float*)d_in[5];
  const float* bv  = (const float*)d_in[6];
  const float* Wg1 = (const float*)d_in[7];
  const float* bg1 = (const float*)d_in[8];
  const float* Wg2 = (const float*)d_in[9];
  const float* bg2 = (const float*)d_in[10];
  const float* Wgr = (const float*)d_in[11];
  const float* bgr = (const float*)d_in[12];
  float* out = (float*)d_out;

  // workspace: wtab f32 [1600] | q bf16 [6400*64] | k bf16 [6400*64]
  //            | vT bf16 [256*6400] (channel-major)
  float*  wtab = (float*)d_ws;
  __bf16* qb   = (__bf16*)((char*)d_ws + 6400);
  __bf16* kb2  = qb + (size_t)6400 * DK;
  __bf16* vtb  = kb2 + (size_t)6400 * DK;

  geo_table_kernel<<<(WGRID * WGRID + 255) / 256, 256, 0, stream>>>(
      Wg1, bg1, Wg2, bg2, Wgr, bgr, wtab);
  qkv_kernel<<<400, 128, 0, stream>>>(x, Wq, bq, Wk, bk, Wv, bv, qb, kb2, vtb);

  size_t smem = (16 * SLAB_LD + WGRID * WGRID + 16 * 8 + 16 + 16) * sizeof(float)
              + (16 * 72) * sizeof(__bf16);
  attn_kernel<<<400, 128, smem, stream>>>(x, qb, kb2, vtb, wtab, out);
}
